// GEN_35115652612103
// MI455X (gfx1250) — matrix operand order repaired and checked by reading, not compiled, not run on hardware
//
#include <hip/hip_runtime.h>
#include <hip/hip_bf16.h>

#define N_NODES 100000
#define N_EDGES 1600000
#define D_IN    128
#define DH      256
#define D_HID   128
#define D_OUTF  64
#define EPS_MSG 1e-7f
#define EPS_BN  1e-5f

typedef __bf16 bf16_t;
typedef __bf16 v16bf __attribute__((ext_vector_type(16)));
typedef __bf16 v8bf  __attribute__((ext_vector_type(8)));
typedef float  v8f   __attribute__((ext_vector_type(8)));

#define CEILDIV(a,b) (((a)+(b)-1)/(b))

// ---------------- zero ----------------
__global__ void k_zero_f32(float* __restrict__ p, size_t n) {
  size_t i = (size_t)blockIdx.x * blockDim.x + threadIdx.x;
  if (i < n) p[i] = 0.f;
}

// ---------------- degree ----------------
__global__ void k_degree(const int* __restrict__ dst, float* __restrict__ deg, int e) {
  int i = blockIdx.x * blockDim.x + threadIdx.x;
  if (i < e) atomicAdd(deg + dst[i], 1.0f);
}

// ---------------- edge scatter: aggr[dst] += relu(h[src]) + eps ----------------
// one wave32 per edge; lane handles 4 contiguous floats (float4 gather)
__global__ void k_scatter(const float* __restrict__ h, const int* __restrict__ src,
                          const int* __restrict__ dst, float* __restrict__ aggr, int e) {
  int edge = (int)(((size_t)blockIdx.x * blockDim.x + threadIdx.x) >> 5);
  int lane = threadIdx.x & 31;
  if (edge >= e) return;
  int s = src[edge], d = dst[edge];
  float4 v = ((const float4*)(h + (size_t)s * D_IN))[lane];
  float* o = aggr + (size_t)d * D_IN + lane * 4;
  atomicAdd(o + 0, fmaxf(v.x, 0.f) + EPS_MSG);
  atomicAdd(o + 1, fmaxf(v.y, 0.f) + EPS_MSG);
  atomicAdd(o + 2, fmaxf(v.z, 0.f) + EPS_MSG);
  atomicAdd(o + 3, fmaxf(v.w, 0.f) + EPS_MSG);
}

// ---------------- A = bf16(aggr/deg + h), row-major [N, 128] ----------------
__global__ void k_aggprep(const float* __restrict__ aggr, const float* __restrict__ h,
                          const float* __restrict__ deg, bf16_t* __restrict__ A, size_t n) {
  size_t i = (size_t)blockIdx.x * blockDim.x + threadIdx.x;
  if (i >= n) return;
  size_t row = i >> 7;  // / D_IN
  float d = fmaxf(deg[row], 1.0f);
  A[i] = (bf16_t)(aggr[i] / d + h[i]);
}

// ---------------- pack fp32 W[K,Nout] into per-lane WMMA B fragments (bf16) ----
// flat index = (((ntile*kblocks + kb)*32 + lane)*16 + e)
// lane col n = ntile*16 + (lane&15); k = kb*32 + 8*(lane>>4) + (e&7) + 16*(e>>3)
__global__ void k_pack_w(const float* __restrict__ W, bf16_t* __restrict__ Bp,
                         int K, int Nout) {
  int idx = blockIdx.x * blockDim.x + threadIdx.x;
  int total = K * Nout;
  if (idx >= total) return;
  int kblocks = K >> 5;
  int e    = idx & 15;
  int lane = (idx >> 4) & 31;
  int rest = idx >> 9;
  int kb    = rest % kblocks;
  int ntile = rest / kblocks;
  int n = ntile * 16 + (lane & 15);
  int k = kb * 32 + ((lane >> 4) << 3) + (e & 7) + ((e >> 3) << 4);
  Bp[idx] = (bf16_t)(W[(size_t)k * Nout + n]);
}

// ---------------- WMMA GEMM: out[M,Nout] = A[M,K] @ W + bias (opt. ELU) -------
// block = 128 (4 waves); each wave owns one 16-row M tile and a 16x64 N strip
__global__ void k_wmma_gemm(const bf16_t* __restrict__ A, const bf16_t* __restrict__ Bp,
                            const float* __restrict__ bias, float* __restrict__ out,
                            int M, int K, int Nout, int do_elu) {
  int wave  = threadIdx.x >> 5;
  int lane  = threadIdx.x & 31;
  int mtile = blockIdx.x * 4 + wave;
  if (mtile >= (M >> 4)) return;
  int ntile0 = blockIdx.y * 4;
  int half   = lane >> 4;
  int mrow   = (mtile << 4) + (lane & 15);
  int kblocks = K >> 5;

  v8f acc[4];
#pragma unroll
  for (int t = 0; t < 4; ++t)
#pragma unroll
    for (int r = 0; r < 8; ++r) acc[t][r] = 0.f;

  const bf16_t* arow = A + (size_t)mrow * K;
  for (int kb = 0; kb < kblocks; ++kb) {
    int k0 = kb << 5;
    // ISA 16-bit A 16x32 layout: lane half h covers K = k0+8h..+8 and k0+16+8h..+8
    v8bf a0 = *(const v8bf*)(arow + k0 + 8 * half);
    v8bf a1 = *(const v8bf*)(arow + k0 + 16 + 8 * half);
    v16bf afrag;
#pragma unroll
    for (int j = 0; j < 8; ++j) { afrag[j] = a0[j]; afrag[8 + j] = a1[j]; }
#pragma unroll
    for (int t = 0; t < 4; ++t) {
      v16bf bfrag = *(const v16bf*)(Bp +
          ((((size_t)(ntile0 + t) * kblocks + kb) * 32 + lane) << 4));
      acc[t] = __builtin_amdgcn_wmma_f32_16x16x32_bf16(
          false, afrag, false, bfrag, (short)0, acc[t], false, false);
    }
  }

  int col_lo = lane & 15;
#pragma unroll
  for (int t = 0; t < 4; ++t) {
    int col = ((ntile0 + t) << 4) + col_lo;
    float b = bias[col];
#pragma unroll
    for (int r = 0; r < 8; ++r) {
      int row = (mtile << 4) + r + 8 * half;  // ISA f32 C/D layout
      float v = acc[t][r] + b;
      if (do_elu) v = v > 0.f ? v : (__expf(v) - 1.f);
      out[(size_t)row * Nout + col] = v;
    }
  }
}

// ---------------- per-column sum / sumsq of z[N, 256] -------------------------
__global__ void k_colstats(const float* __restrict__ z, float* __restrict__ sums, int M) {
  int col  = threadIdx.x;           // blockDim.x == 256 == DH
  int row0 = blockIdx.x * 128;
  int rend = row0 + 128; if (rend > M) rend = M;
  float s = 0.f, sq = 0.f;
  for (int r = row0; r < rend; ++r) {
    float v = z[(size_t)r * DH + col];
    s += v; sq += v * v;
  }
  atomicAdd(sums + col, s);
  atomicAdd(sums + DH + col, sq);
}

// ---------------- BN + ReLU + bf16 pack ---------------------------------------
__global__ void k_bnpack(const float* __restrict__ z, const float* __restrict__ sums,
                         const float* __restrict__ g, const float* __restrict__ be,
                         bf16_t* __restrict__ A2, size_t n) {
  size_t i = (size_t)blockIdx.x * blockDim.x + threadIdx.x;
  if (i >= n) return;
  int col = (int)(i & (DH - 1));
  const float invN = 1.f / (float)N_NODES;
  float mean = sums[col] * invN;
  float var  = sums[DH + col] * invN - mean * mean;
  float v = (z[i] - mean) * rsqrtf(var + EPS_BN) * g[col] + be[col];
  A2[i] = (bf16_t)fmaxf(v, 0.f);
}

extern "C" void kernel_launch(void* const* d_in, const int* in_sizes, int n_in,
                              void* d_out, int out_size, void* d_ws, size_t ws_size,
                              hipStream_t stream) {
  const float* x    = (const float*)d_in[0];
  const float* W1s  = (const float*)d_in[1];   // [3,128,256]
  const float* b1s  = (const float*)d_in[2];   // [3,256]
  const float* g1s  = (const float*)d_in[3];
  const float* be1s = (const float*)d_in[4];
  const float* W2s  = (const float*)d_in[5];   // [3,256,128]
  const float* b2s  = (const float*)d_in[6];   // [3,128]
  const float* Wo1  = (const float*)d_in[7];   // [128,256]
  const float* bo1  = (const float*)d_in[8];
  const float* go1  = (const float*)d_in[9];
  const float* beo1 = (const float*)d_in[10];
  const float* Wo2  = (const float*)d_in[11];  // [256,64]
  const float* bo2  = (const float*)d_in[12];
  const int*   ei   = (const int*)d_in[13];    // [2, E]
  const int* src = ei;
  const int* dst = ei + N_EDGES;

  // ---- workspace carve (256B aligned) ----
  char* w = (char*)d_ws;
  auto carve = [&](size_t bytes) {
    char* p = w; w += (bytes + 255) & ~(size_t)255; return p;
  };
  float*  deg  = (float*)carve((size_t)N_NODES * 4);
  float*  aggr = (float*)carve((size_t)N_NODES * D_IN * 4);
  float*  hbuf = (float*)carve((size_t)N_NODES * D_IN * 4);
  float*  z    = (float*)carve((size_t)N_NODES * DH * 4);
  bf16_t* Abf  = (bf16_t*)carve((size_t)N_NODES * D_IN * 2);
  float*  sums = (float*)carve(2 * DH * 4);
  bf16_t* W1p  = (bf16_t*)carve((size_t)D_IN * DH * 2);
  bf16_t* W2p  = (bf16_t*)carve((size_t)DH * D_HID * 2);
  bf16_t* A2bf = (bf16_t*)aggr;  // alias: aggr dead after k_aggprep, reused for BN output

  // ---- degree (once) ----
  k_zero_f32<<<CEILDIV(N_NODES, 256), 256, 0, stream>>>(deg, (size_t)N_NODES);
  k_degree<<<CEILDIV(N_EDGES, 256), 256, 0, stream>>>(dst, deg, N_EDGES);

  const float* h = x;
  const size_t nA  = (size_t)N_NODES * D_IN;
  const size_t nZ  = (size_t)N_NODES * DH;
  const int mtiles = N_NODES / 16;

  for (int layer = 0; layer < 4; ++layer) {
    const bool hidden = (layer < 3);
    const float* W1 = hidden ? (W1s  + (size_t)layer * D_IN * DH) : Wo1;
    const float* b1 = hidden ? (b1s  + (size_t)layer * DH)        : bo1;
    const float* g1 = hidden ? (g1s  + (size_t)layer * DH)        : go1;
    const float* be = hidden ? (be1s + (size_t)layer * DH)        : beo1;
    const float* W2 = hidden ? (W2s  + (size_t)layer * DH * D_HID): Wo2;
    const float* b2 = hidden ? (b2s  + (size_t)layer * D_HID)     : bo2;
    const int nout2 = hidden ? D_HID : D_OUTF;
    float* out2     = hidden ? hbuf  : (float*)d_out;

    // 1) aggregate messages
    k_zero_f32<<<CEILDIV(nA, (size_t)256), 256, 0, stream>>>(aggr, nA);
    k_scatter<<<CEILDIV((size_t)N_EDGES * 32, (size_t)256), 256, 0, stream>>>(
        h, src, dst, aggr, N_EDGES);

    // 2) A = bf16(aggr/deg + h)
    k_aggprep<<<CEILDIV(nA, (size_t)256), 256, 0, stream>>>(aggr, h, deg, Abf, nA);

    // 3) z = A @ W1 + b1   [N,128]x[128,256]
    k_pack_w<<<CEILDIV(D_IN * DH, 256), 256, 0, stream>>>(W1, W1p, D_IN, DH);
    {
      dim3 grid(CEILDIV(mtiles, 4), DH / 64);
      k_wmma_gemm<<<grid, 128, 0, stream>>>(Abf, W1p, b1, z, N_NODES, D_IN, DH, 0);
    }

    // 4) batch-norm stats + BN/ReLU/pack
    k_zero_f32<<<CEILDIV((size_t)(2 * DH), (size_t)256), 256, 0, stream>>>(sums, 2 * DH);
    k_colstats<<<CEILDIV(N_NODES, 128), DH, 0, stream>>>(z, sums, N_NODES);
    k_bnpack<<<CEILDIV(nZ, (size_t)256), 256, 0, stream>>>(z, sums, g1, be, A2bf, nZ);

    // 5) out = relu(bn(z)) @ W2 + b2   (+ ELU for hidden layers)
    k_pack_w<<<CEILDIV(DH * nout2, 256), 256, 0, stream>>>(W2, W2p, DH, nout2);
    {
      dim3 grid(CEILDIV(mtiles, 4), nout2 / 64);
      k_wmma_gemm<<<grid, 128, 0, stream>>>(A2bf, W2p, b2, out2, N_NODES, DH, nout2,
                                            hidden ? 1 : 0);
    }

    h = hbuf;
  }
}
